// CasualSelfAttention_69148973466281
// MI455X (gfx1250) — matrix operand order repaired and checked by reading, not compiled, not run on hardware
//
#include <hip/hip_runtime.h>

// ---------------------------------------------------------------------------
// CDNA5 (gfx1250) full self-attention pipeline in bf16 WMMA, f32 accumulate,
// with async global->LDS staging (ASYNCcnt) and double buffering.
// B=2, T=2048, C=1024, H=16, hd=64.  scale = C^-0.5 = 1/32.
// ---------------------------------------------------------------------------

typedef __bf16 bf16;
typedef __attribute__((ext_vector_type(8)))  float v8f;
typedef __attribute__((ext_vector_type(4)))  int   v4i;
typedef __attribute__((ext_vector_type(8)))  bf16  bf16x8;
typedef __attribute__((ext_vector_type(16))) bf16  bf16x16;

// Pointer-to-addrspace-qualified-int4 types for the async LDS builtins.
typedef __attribute__((address_space(1))) v4i* gas_v4i_ptr;
typedef __attribute__((address_space(3))) v4i* las_v4i_ptr;

#define WMMA_BF16(a, b, c) \
  __builtin_amdgcn_wmma_f32_16x16x32_bf16(false, (a), false, (b), (short)0, (c), false, false)

#ifndef __has_builtin
#define __has_builtin(x) 0
#endif
#if __has_builtin(__builtin_amdgcn_global_load_async_to_lds_b128) && \
    __has_builtin(__builtin_amdgcn_s_wait_asynccnt)
#define HAS_ASYNC 1
#else
#define HAS_ASYNC 0
#endif

// One 16-byte global -> LDS copy.  Async path uses GLOBAL_LOAD_ASYNC_TO_LDS_B128
// (no VGPR data, tracked by ASYNCcnt); fallback is load+ds_store.
static __device__ inline void async_copy16(const bf16* g, bf16* l) {
#if HAS_ASYNC
  __builtin_amdgcn_global_load_async_to_lds_b128(
      (gas_v4i_ptr)(g), (las_v4i_ptr)(l), 0, 0);
#else
  *(uint4*)l = *(const uint4*)g;
#endif
}

static __device__ inline void async_wait0() {
#if HAS_ASYNC
  __builtin_amdgcn_s_wait_asynccnt(0);
#endif
}

// A-fragment (16x32 bf16): lane<16 holds K 0..15 of its row,
// lane>=16 holds K 16..31.  `slice` points at the row's K=0 element.
static __device__ inline bf16x16 ldA(const bf16* slice, int hi) {
  return *(const bf16x16*)(slice + hi * 16);
}

// B-fragment (32x16 bf16): lane<16 holds K 0..15 of its column, lane>=16 K 16..31.
static __device__ inline bf16x16 ldB(const bf16* slice, int hi) {
  return *(const bf16x16*)(slice + hi * 16);
}

// ---------------------------------------------------------------------------
// fp32 -> bf16 bulk convert (8 elements / thread)
// ---------------------------------------------------------------------------
__global__ __launch_bounds__(256) void cvt_bf16_kernel(const float* __restrict__ in,
                                                       bf16* __restrict__ out, int n) {
  int i = (blockIdx.x * blockDim.x + threadIdx.x) * 8;
  if (i >= n) return;
  float4 a = *(const float4*)(in + i);
  float4 b = *(const float4*)(in + i + 4);
  bf16x8 v;
  v[0] = (bf16)a.x; v[1] = (bf16)a.y; v[2] = (bf16)a.z; v[3] = (bf16)a.w;
  v[4] = (bf16)b.x; v[5] = (bf16)b.y; v[6] = (bf16)b.z; v[7] = (bf16)b.w;
  *(bf16x8*)(out + i) = v;
}

// ---------------------------------------------------------------------------
// GEMM1: qkv = x @ w_attn^T + b_attn, scattered into per-head Q/K [B,H,T,64]
// and transposed V [B,H,64,T].  M=4096, N=3072, K=1024.
// Block tile 128x128, BK=64 double-buffered async, 8 waves of 32x64.
// ---------------------------------------------------------------------------
__global__ __launch_bounds__(256) void qkv_gemm_kernel(const bf16* __restrict__ xb,
                                                       const bf16* __restrict__ wb,
                                                       const float* __restrict__ bias,
                                                       bf16* __restrict__ Qb,
                                                       bf16* __restrict__ Kb,
                                                       bf16* __restrict__ Vt) {
  const int K = 1024, BK = 64;
  __shared__ bf16 As[2][128][BK];  // 32 KB
  __shared__ bf16 Bs[2][128][BK];  // 32 KB
  const int tid = threadIdx.x;
  const int mbase = blockIdx.y * 128;
  const int nbase = blockIdx.x * 128;
  const int wave = tid >> 5, lane = tid & 31;
  const int l16 = lane & 15, hi = lane >> 4;
  const int wm = wave & 3, wn = wave >> 2;

  auto issue = [&](int kb, int buf) {
#pragma unroll
    for (int i = 0; i < 4; i++) {
      int c = tid + i * 256;                 // 1024 16B chunks per tile
      int row = c >> 3, ko = (c & 7) * 8;    // 8 chunks per 64-elem row
      async_copy16(&xb[(size_t)(mbase + row) * K + kb + ko], &As[buf][0][0] + c * 8);
      async_copy16(&wb[(size_t)(nbase + row) * K + kb + ko], &Bs[buf][0][0] + c * 8);
    }
  };

  v8f acc[2][4];
  v8f zero = {};
  for (int i = 0; i < 2; i++)
    for (int j = 0; j < 4; j++) acc[i][j] = zero;

  issue(0, 0);
  async_wait0();
  __syncthreads();

  for (int kb = 0; kb < K; kb += BK) {
    int buf = (kb >> 6) & 1;
    if (kb + BK < K) issue(kb + BK, buf ^ 1);
#pragma unroll
    for (int ks = 0; ks < 2; ks++) {
      bf16x16 af[2], bfm[4];
#pragma unroll
      for (int mi = 0; mi < 2; mi++) af[mi] = ldA(&As[buf][wm * 32 + mi * 16 + l16][ks * 32], hi);
#pragma unroll
      for (int ni = 0; ni < 4; ni++) bfm[ni] = ldB(&Bs[buf][wn * 64 + ni * 16 + l16][ks * 32], hi);
#pragma unroll
      for (int mi = 0; mi < 2; mi++)
#pragma unroll
        for (int ni = 0; ni < 4; ni++) acc[mi][ni] = WMMA_BF16(af[mi], bfm[ni], acc[mi][ni]);
    }
    async_wait0();
    __syncthreads();
  }

  // Epilogue: scatter column n into head-interleaved Q/K/V (h = n/192, j = n%192).
  // V goes out transposed: [b,h][dh][t].
#pragma unroll
  for (int mi = 0; mi < 2; mi++)
#pragma unroll
    for (int ni = 0; ni < 4; ni++) {
      int n = nbase + wn * 64 + ni * 16 + l16;
      int h = n / 192, j = n % 192;
      int sel = j >> 6, dh = j & 63;
      bf16* dst = (sel == 0) ? Qb : (sel == 1 ? Kb : Vt);
      float bv = bias[n];
#pragma unroll
      for (int r = 0; r < 8; r++) {
        int m = mbase + wm * 32 + mi * 16 + r + 8 * hi;  // C-frag: row = r + 8*hi
        int b = m >> 11, t = m & 2047;
        size_t bh = (size_t)(b * 16 + h);
        size_t idx = (sel == 2) ? ((bh * 64 + dh) * 2048 + t)
                                : ((bh * 2048 + t) * 64 + dh);
        dst[idx] = (bf16)(acc[mi][ni][r] + bv);
      }
    }
}

// ---------------------------------------------------------------------------
// Flash attention: 1 block = 4 waves = 64 q rows of one (b,h). Key tiles of 32,
// K/V double-buffered via async LDS copies.  V arrives pre-transposed [dh][t].
// ---------------------------------------------------------------------------
__global__ __launch_bounds__(128) void attn_kernel(const bf16* __restrict__ Qb,
                                                   const bf16* __restrict__ Kb,
                                                   const bf16* __restrict__ Vt,
                                                   bf16* __restrict__ yb) {
  const int T = 2048, HD = 64;
  __shared__ bf16 Qs[64][64];        // 8 KB
  __shared__ bf16 Ks[2][32][64];     // 8 KB, [key][dh]
  __shared__ bf16 Vts[2][64][32];    // 8 KB, [dh][key]
  __shared__ bf16 Ps[4][16][32];     // 4 KB, per-wave P staging

  const int tid = threadIdx.x;
  const int wave = tid >> 5, lane = tid & 31;
  const int l16 = lane & 15, hi = lane >> 4;
  const int bh = blockIdx.y;           // b*16 + h
  const int qbase = blockIdx.x * 64;
  const bf16* Qh = Qb + (size_t)bh * T * HD;
  const bf16* Kh = Kb + (size_t)bh * T * HD;
  const bf16* Vth = Vt + (size_t)bh * HD * T;   // [dh][t]

  auto issue_kv = [&](int kb, int buf) {
#pragma unroll
    for (int i = 0; i < 2; i++) {
      int c = tid + i * 128;               // 256 chunks each
      int kr = c >> 2, ko = (c & 3) * 8;   // K tile: 64 keys x 4 chunks (dh 0..31)
      async_copy16(&Kh[(size_t)(kb + kr) * HD + ko], &Ks[buf][0][0] + c * 8);
      int dh = c >> 2, to = (c & 3) * 8;   // V tile: 64 dh rows x 4 chunks of keys
      async_copy16(&Vth[(size_t)dh * T + kb + to], &Vts[buf][0][0] + c * 8);
    }
  };

  // Cooperative Q tile load (64x64), once
#pragma unroll
  for (int i = 0; i < 2; i++) {
    int c = tid * 2 + i;
    int row = c >> 2, seg = c & 3;
    *(bf16x16*)&Qs[row][seg * 16] = *(const bf16x16*)&Qh[(size_t)(qbase + row) * HD + seg * 16];
  }

  float rm[8], rl[8];
#pragma unroll
  for (int r = 0; r < 8; r++) { rm[r] = -3.0e38f; rl[r] = 0.0f; }
  v8f o[4];
  v8f zero = {};
#pragma unroll
  for (int i = 0; i < 4; i++) o[i] = zero;

  const float scale = 0.03125f;  // C^-0.5 = 1/32 (reference scales by model dim)

  issue_kv(0, 0);
  async_wait0();
  __syncthreads();

  for (int kb = 0; kb < T; kb += 32) {
    int buf = (kb >> 5) & 1;
    if (kb + 32 < T) issue_kv(kb + 32, buf ^ 1);

    // S = Q Kt for 16 q rows x 32 keys, head dim 64 in 2 k-steps
    v8f s0 = zero, s1 = zero;
#pragma unroll
    for (int ks = 0; ks < 2; ks++) {
      bf16x16 aq  = ldA(&Qs[wave * 16 + l16][ks * 32], hi);
      bf16x16 bk0 = ldB(&Ks[buf][l16][ks * 32], hi);        // keys 0..15
      bf16x16 bk1 = ldB(&Ks[buf][16 + l16][ks * 32], hi);   // keys 16..31
      s0 = WMMA_BF16(aq, bk0, s0);
      s1 = WMMA_BF16(aq, bk1, s1);
    }

    // Online softmax.  C-frag: vgpr r holds row r + 8*hi, lane%16 = column.
#pragma unroll
    for (int r = 0; r < 8; r++) {
      float a = s0[r] * scale, b = s1[r] * scale;
      float tmax = fmaxf(a, b);
#pragma unroll
      for (int m = 1; m < 16; m <<= 1) tmax = fmaxf(tmax, __shfl_xor(tmax, m, 32));
      float nm = fmaxf(rm[r], tmax);
      float alpha = __expf(rm[r] - nm);
      float p0 = __expf(a - nm);
      float p1 = __expf(b - nm);
      float ts = p0 + p1;
#pragma unroll
      for (int m = 1; m < 16; m <<= 1) ts += __shfl_xor(ts, m, 32);
      rl[r] = rl[r] * alpha + ts;
      rm[r] = nm;
#pragma unroll
      for (int ni = 0; ni < 4; ni++) o[ni][r] *= alpha;
      // Stage P to LDS (C-layout -> row-major) for A-fragment reload
      Ps[wave][r + 8 * hi][l16]      = (bf16)p0;
      Ps[wave][r + 8 * hi][16 + l16] = (bf16)p1;
    }

    // O += P @ V   (A = P 16x32 over keys, B = Vt columns = dh)
    bf16x16 ap = ldA(&Ps[wave][l16][0], hi);
#pragma unroll
    for (int ni = 0; ni < 4; ni++) {
      bf16x16 bv = ldB(&Vts[buf][ni * 16 + l16][0], hi);
      o[ni] = WMMA_BF16(ap, bv, o[ni]);
    }
    async_wait0();
    __syncthreads();  // protect Ks/Vts buf^1 before next iteration reads it
  }

  // Normalize and write y (bf16) in [B,T,C] layout, c = h*64 + dh
  const int b = bh >> 4, h = bh & 15;
#pragma unroll
  for (int r = 0; r < 8; r++) {
    float inv = 1.0f / rl[r];
    int t = qbase + wave * 16 + r + 8 * hi;
#pragma unroll
    for (int ni = 0; ni < 4; ni++) {
      int c = h * 64 + ni * 16 + l16;
      yb[(size_t)(b * T + t) * 1024 + c] = (bf16)(o[ni][r] * inv);
    }
  }
}

// ---------------------------------------------------------------------------
// GEMM2: out = y @ w_proj^T + b_proj (fp32 out).  M=4096, N=1024, K=1024.
// ---------------------------------------------------------------------------
__global__ __launch_bounds__(256) void proj_gemm_kernel(const bf16* __restrict__ yb,
                                                        const bf16* __restrict__ wpb,
                                                        const float* __restrict__ bias,
                                                        float* __restrict__ out) {
  const int K = 1024, BK = 64;
  __shared__ bf16 As[2][128][BK];
  __shared__ bf16 Bs[2][128][BK];
  const int tid = threadIdx.x;
  const int mbase = blockIdx.y * 128;
  const int nbase = blockIdx.x * 128;
  const int wave = tid >> 5, lane = tid & 31;
  const int l16 = lane & 15, hi = lane >> 4;
  const int wm = wave & 3, wn = wave >> 2;

  auto issue = [&](int kb, int buf) {
#pragma unroll
    for (int i = 0; i < 4; i++) {
      int c = tid + i * 256;
      int row = c >> 3, ko = (c & 7) * 8;
      async_copy16(&yb[(size_t)(mbase + row) * K + kb + ko], &As[buf][0][0] + c * 8);
      async_copy16(&wpb[(size_t)(nbase + row) * K + kb + ko], &Bs[buf][0][0] + c * 8);
    }
  };

  v8f acc[2][4];
  v8f zero = {};
  for (int i = 0; i < 2; i++)
    for (int j = 0; j < 4; j++) acc[i][j] = zero;

  issue(0, 0);
  async_wait0();
  __syncthreads();

  for (int kb = 0; kb < K; kb += BK) {
    int buf = (kb >> 6) & 1;
    if (kb + BK < K) issue(kb + BK, buf ^ 1);
#pragma unroll
    for (int ks = 0; ks < 2; ks++) {
      bf16x16 af[2], bfm[4];
#pragma unroll
      for (int mi = 0; mi < 2; mi++) af[mi] = ldA(&As[buf][wm * 32 + mi * 16 + l16][ks * 32], hi);
#pragma unroll
      for (int ni = 0; ni < 4; ni++) bfm[ni] = ldB(&Bs[buf][wn * 64 + ni * 16 + l16][ks * 32], hi);
#pragma unroll
      for (int mi = 0; mi < 2; mi++)
#pragma unroll
        for (int ni = 0; ni < 4; ni++) acc[mi][ni] = WMMA_BF16(af[mi], bfm[ni], acc[mi][ni]);
    }
    async_wait0();
    __syncthreads();
  }

#pragma unroll
  for (int mi = 0; mi < 2; mi++)
#pragma unroll
    for (int ni = 0; ni < 4; ni++) {
      int n = nbase + wn * 64 + ni * 16 + l16;
      float bv = bias[n];
#pragma unroll
      for (int r = 0; r < 8; r++) {
        int m = mbase + wm * 32 + mi * 16 + r + 8 * hi;
        out[(size_t)m * 1024 + n] = acc[mi][ni][r] + bv;
      }
    }
}

// ---------------------------------------------------------------------------
// Host launcher
// ---------------------------------------------------------------------------
extern "C" void kernel_launch(void* const* d_in, const int* in_sizes, int n_in,
                              void* d_out, int out_size, void* d_ws, size_t ws_size,
                              hipStream_t stream) {
  const float* x      = (const float*)d_in[0];
  const float* w_attn = (const float*)d_in[1];
  const float* b_attn = (const float*)d_in[2];
  const float* w_proj = (const float*)d_in[3];
  const float* b_proj = (const float*)d_in[4];
  float* out = (float*)d_out;

  const int B = 2, T = 2048, C = 1024, H = 16, HD = 64;
  const size_t M = (size_t)B * T;  // 4096

  char* w = (char*)d_ws;
  bf16* xb  = (bf16*)w; w += M * C * sizeof(bf16);                   // 8 MB
  bf16* wb  = (bf16*)w; w += (size_t)3 * C * C * sizeof(bf16);       // 6 MB
  bf16* wpb = (bf16*)w; w += (size_t)C * C * sizeof(bf16);           // 2 MB
  bf16* Qb  = (bf16*)w; w += (size_t)B * H * T * HD * sizeof(bf16);  // 8 MB
  bf16* Kb  = (bf16*)w; w += (size_t)B * H * T * HD * sizeof(bf16);  // 8 MB
  bf16* Vt  = (bf16*)w; w += (size_t)B * H * HD * T * sizeof(bf16);  // 8 MB (transposed)
  bf16* yb  = (bf16*)w; w += M * C * sizeof(bf16);                   // 8 MB

  // 1) fp32 -> bf16 converts
  cvt_bf16_kernel<<<(int)(M * C / 2048), 256, 0, stream>>>(x, xb, (int)(M * C));
  cvt_bf16_kernel<<<(3 * C * C) / 2048, 256, 0, stream>>>(w_attn, wb, 3 * C * C);
  cvt_bf16_kernel<<<(C * C) / 2048, 256, 0, stream>>>(w_proj, wpb, C * C);

  // 2) QKV projection + head scatter (V transposed)
  qkv_gemm_kernel<<<dim3(24, 32), 256, 0, stream>>>(xb, wb, b_attn, Qb, Kb, Vt);

  // 3) flash attention per (b,h), 64 q rows per block
  attn_kernel<<<dim3(T / 64, B * H), 128, 0, stream>>>(Qb, Kb, Vt, yb);

  // 4) output projection (fp32 out)
  proj_gemm_kernel<<<dim3(8, 32), 256, 0, stream>>>(yb, wpb, b_proj, out);
}
